// Diffuser_38362647888260
// MI455X (gfx1250) — matrix operand order repaired and checked by reading, not compiled, not run on hardware
//
#include <hip/hip_runtime.h>

// ---------------------------------------------------------------------------
// Types for CDNA5 WMMA (wave32): bf16 16x16x32, f32 accumulate
// ---------------------------------------------------------------------------
typedef __bf16 bf16_t;
typedef bf16_t bf16x16 __attribute__((ext_vector_type(16)));
typedef float  floatx8 __attribute__((ext_vector_type(8)));

union FragBF4 { bf16x16 v; uint4 q[2]; };
union FragF   { floatx8 v; float f[8]; };
union BFBits  { bf16_t h; unsigned short s; };

// Native f32 -> bf16 (backend uses v_cvt/_fma_mix*_bf16)
__device__ __forceinline__ unsigned short f2bf(float f) {
  BFBits b; b.h = (bf16_t)f;
  return b.s;
}
__device__ __forceinline__ unsigned int pack2bf(float lo, float hi) {
  return ((unsigned int)f2bf(hi) << 16) | (unsigned int)f2bf(lo);
}

// A fragment: 16x32 bf16, row-major in LDS, stride lda (elements).
// ISA 7.12.2: lane m=l&15, half=l>>4; VGPRs 0-3 hold K={0..7}+8*half (pairs),
// VGPRs 4-7 hold K={16..23}+8*half. Each run is 16 contiguous bytes -> b128.
__device__ __forceinline__ bf16x16 load_frag_a(const unsigned short* lds,
                                               int rowbase, int k0, int lda) {
  int lane = threadIdx.x & 31;
  int m = lane & 15, half = lane >> 4;
  const uint4* p = (const uint4*)(lds + (rowbase + m) * lda + k0 + 8*half);
  FragBF4 fr;
  fr.q[0] = p[0];     // K = 8*half .. 8*half+7
  fr.q[1] = p[2];     // K = 16+8*half .. 23+8*half
  return fr.v;
}

// B fragment: 32x16 bf16. LDS holds B transposed ("Bt"): Bt[col][k], stride ldbt.
// Same K order as A: VGPRs 0-3 hold K={0..7}+8*half, VGPRs 4-7 hold K={16..23}+8*half.
// Two runs of 8 contiguous K-shorts -> two b128 loads.
__device__ __forceinline__ bf16x16 load_frag_b(const unsigned short* lds,
                                               int colbase, int k0, int ldbt) {
  int lane = threadIdx.x & 31;
  int n = lane & 15, half = lane >> 4;
  const uint4* p = (const uint4*)(lds + (colbase + n) * ldbt + k0 + 8*half);
  FragBF4 fr;
  fr.q[0] = p[0];     // K = 8*half .. 8*half+7
  fr.q[1] = p[2];     // K = 16+8*half .. 23+8*half
  return fr.v;
}

__device__ __forceinline__ floatx8 wmma_bf16(bf16x16 a, bf16x16 b, floatx8 c) {
  return __builtin_amdgcn_wmma_f32_16x16x32_bf16(false, a, false, b,
                                                 (short)0, c, false, false);
}

// ---------------------------------------------------------------------------
// Generic tiled bf16 GEMM: C(MxNc) = A_f32(MxK) x B_bf16(KxNc)
// 64x64 tile, K-step 32, 128 threads (4 waves, 2x2 wave grid).
// All M/Nc are multiples of 64 -> no bounds checks.
// MODE 0: scatter to k_bf (R,H,N,hd) bf16
// MODE 1: split/scatter to q_bf, v_bf (B,H,N,hd) bf16
// MODE 2: fp32 out + bias
// ---------------------------------------------------------------------------
template<int MODE>
__global__ __launch_bounds__(128) void gemm_bf16(
    const float* __restrict__ A, const unsigned short* __restrict__ Bw,
    int M, int Nc, int K,
    unsigned short* __restrict__ out0, unsigned short* __restrict__ out1,
    float* __restrict__ outf, const float* __restrict__ bias)
{
  __shared__ alignas(16) unsigned short sA[64*32];   // [row][k]
  __shared__ alignas(16) unsigned short sB[64*32];   // Bt: [col][k]
  int tid = threadIdx.x;
  int row0 = blockIdx.y * 64;
  int col0 = blockIdx.x * 64;
  int wid = tid >> 5;
  int wm = wid >> 1, wn = wid & 1;

  FragF acc[2][2];
#pragma unroll
  for (int i = 0; i < 2; ++i)
#pragma unroll
    for (int j = 0; j < 2; ++j)
#pragma unroll
      for (int e = 0; e < 8; ++e) acc[i][j].f[e] = 0.0f;

  for (int k0 = 0; k0 < K; k0 += 32) {
    __syncthreads();
    // A tile: 64x32 fp32, b128 loads + packed bf16 stores (8B) to LDS
    {
      const float4* A4 = (const float4*)A;
      uint2* sA2 = (uint2*)sA;
#pragma unroll
      for (int i = 0; i < 4; ++i) {
        int idx = tid + i*128;                 // 512 float4 per tile
        int r = idx >> 3, c4 = idx & 7;        // row, 16B-chunk within row
        float4 v = A4[(((size_t)(row0 + r))*K + k0) / 4 + c4];
        uint2 pk;
        pk.x = pack2bf(v.x, v.y);
        pk.y = pack2bf(v.z, v.w);
        sA2[idx] = pk;                         // row-major: r*8 + c4
      }
    }
    // B tile: 32x64 bf16, dword loads (2 cols) + transposed b16 stores
    {
      const unsigned int* B4 = (const unsigned int*)Bw;
#pragma unroll
      for (int i = 0; i < 8; ++i) {
        int idx = tid + i*128;                 // 1024 dwords per tile
        int kk = idx >> 5, c2 = idx & 31;      // k row, col-pair
        unsigned int val = B4[(((size_t)(k0 + kk))*Nc + col0 + c2*2) >> 1];
        sB[(c2*2    )*32 + kk] = (unsigned short)val;
        sB[(c2*2 + 1)*32 + kk] = (unsigned short)(val >> 16);
      }
    }
    __syncthreads();

    bf16x16 bfr[2];
    bfr[0] = load_frag_b(sB, wn*32,      0, 32);
    bfr[1] = load_frag_b(sB, wn*32 + 16, 0, 32);
#pragma unroll
    for (int ti = 0; ti < 2; ++ti) {
      bf16x16 a = load_frag_a(sA, wm*32 + ti*16, 0, 32);
#pragma unroll
      for (int tj = 0; tj < 2; ++tj)
        acc[ti][tj].v = wmma_bf16(a, bfr[tj], acc[ti][tj].v);
    }
  }

  int lane = tid & 31;
  int nl = lane & 15, half = lane >> 4;
#pragma unroll
  for (int ti = 0; ti < 2; ++ti)
#pragma unroll
    for (int tj = 0; tj < 2; ++tj)
#pragma unroll
      for (int i = 0; i < 8; ++i) {
        int gm = row0 + wm*32 + ti*16 + i + half*8;
        int gc = col0 + wn*32 + tj*16 + nl;
        float v = acc[ti][tj].f[i];
        if (MODE == 0) {                      // k: (R,N)x(H,hd) -> (R,H,N,hd)
          int r = gm >> 8, n = gm & 255;
          int h = gc >> 6, hd = gc & 63;
          out0[(((size_t)r*12 + h)*256 + n)*64 + hd] = f2bf(v);
        } else if (MODE == 1) {               // qv: split 2x(H,hd)
          int b = gm >> 8, n = gm & 255;
          int sel = gc >= 768;
          int rem = sel ? gc - 768 : gc;
          int h = rem >> 6, hd = rem & 63;
          size_t dst = (((size_t)b*12 + h)*256 + n)*64 + hd;
          (sel ? out1 : out0)[dst] = f2bf(v);
        } else {                              // proj: fp32 out + bias
          outf[(size_t)gm*Nc + gc] = v + bias[gc];
        }
      }
}

// ---------------------------------------------------------------------------
// Fused attention: one workgroup per (b,r,h); 8 waves; 256 KB LDS resident.
// S = q k^T * 0.125  (bf16, 128KB LDS)
// T = S S^T          (bf16, 128KB LDS; region reused from q/k)
// Y = T v            -> atomicAdd(x_acc) with torch raw-reshape permutation
// ---------------------------------------------------------------------------
__global__ __launch_bounds__(256) void attn_fused(
    const unsigned short* __restrict__ qg, const unsigned short* __restrict__ kg,
    const unsigned short* __restrict__ vg, float* __restrict__ xacc)
{
  __shared__ alignas(16) unsigned short sS[65536];   // 128 KB: S, later v^T
  __shared__ alignas(16) unsigned short sT[65536];   // 128 KB: q|k, later T

  int idx = blockIdx.x;
  int b  = idx / 120;            // R*H = 120
  int rh = idx - b*120;
  int r  = rh / 12;
  int h  = rh - r*12;
  int tid = threadIdx.x;
  int w   = tid >> 5;
  int lane = tid & 31, nl = lane & 15, half = lane >> 4;

  const unsigned short* q = qg + (((size_t)b*12 + h) * 256) * 64;
  const unsigned short* k = kg + (((size_t)r*12 + h) * 256) * 64;
  const unsigned short* v = vg + (((size_t)b*12 + h) * 256) * 64;

  unsigned short* sQ = sT;            // 256x64
  unsigned short* sK = sT + 16384;    // 256x64
  {
    const uint4* q4 = (const uint4*)q;
    const uint4* k4 = (const uint4*)k;
    uint4* sQ4 = (uint4*)sQ;
    uint4* sK4 = (uint4*)sK;
#pragma unroll
    for (int i = 0; i < 8; ++i) {            // 2048 uint4 each
      int j = tid + i*256;
      sQ4[j] = q4[j];
      sK4[j] = k4[j];
    }
  }
  __syncthreads();

  // Phase 1: S = q k^T * scale   (256x256, 256 tiles / 8 waves)
  for (int it = 0; it < 32; ++it) {
    int t = w + (it << 3);
    int tm = t >> 4, tn = t & 15;
    FragF acc;
#pragma unroll
    for (int e = 0; e < 8; ++e) acc.f[e] = 0.0f;
#pragma unroll
    for (int kk = 0; kk < 2; ++kk) {
      bf16x16 a  = load_frag_a(sQ, tm*16, kk*32, 64);
      bf16x16 bb = load_frag_b(sK, tn*16, kk*32, 64);  // Bt[m][d] = k[m][d]
      acc.v = wmma_bf16(a, bb, acc.v);
    }
#pragma unroll
    for (int i = 0; i < 8; ++i)
      sS[(tm*16 + i + half*8)*256 + tn*16 + nl] = f2bf(acc.f[i] * 0.125f);
  }
  __syncthreads();

  // Phase 2: T = S S^T   (overwrites dead q/k region)
  for (int it = 0; it < 32; ++it) {
    int t = w + (it << 3);
    int tm = t >> 4, tn = t & 15;
    FragF acc;
#pragma unroll
    for (int e = 0; e < 8; ++e) acc.f[e] = 0.0f;
#pragma unroll
    for (int kk = 0; kk < 8; ++kk) {
      bf16x16 a  = load_frag_a(sS, tm*16, kk*32, 256);
      bf16x16 bb = load_frag_b(sS, tn*16, kk*32, 256); // Bt[p][m] = S[p][m]
      acc.v = wmma_bf16(a, bb, acc.v);
    }
#pragma unroll
    for (int i = 0; i < 8; ++i)
      sT[(tm*16 + i + half*8)*256 + tn*16 + nl] = f2bf(acc.f[i]);
  }
  __syncthreads();

  // v^T into dead S region: vT[hd][p]; b128 global loads, b16 LDS scatter
  {
    const uint4* v4 = (const uint4*)v;
#pragma unroll
    for (int i = 0; i < 8; ++i) {
      int j = tid + i*256;          // 2048 uint4 = full 256x64 tile
      int p  = j >> 3;              // row (token)
      int h8 = (j & 7) * 8;         // hd base of this 8-element chunk
      uint4 val = v4[j];
      sS[(h8+0)*256 + p] = (unsigned short)val.x;
      sS[(h8+1)*256 + p] = (unsigned short)(val.x >> 16);
      sS[(h8+2)*256 + p] = (unsigned short)val.y;
      sS[(h8+3)*256 + p] = (unsigned short)(val.y >> 16);
      sS[(h8+4)*256 + p] = (unsigned short)val.z;
      sS[(h8+5)*256 + p] = (unsigned short)(val.z >> 16);
      sS[(h8+6)*256 + p] = (unsigned short)val.w;
      sS[(h8+7)*256 + p] = (unsigned short)(val.w >> 16);
    }
  }
  __syncthreads();

  // Phase 3: Y = T v ; scatter-mean into x_acc (raw (B,R,H,N,hd)->(B,R,N,C))
  for (int it = 0; it < 8; ++it) {
    int t = w + (it << 3);
    int tm = t >> 2, thd = t & 3;
    FragF acc;
#pragma unroll
    for (int e = 0; e < 8; ++e) acc.f[e] = 0.0f;
#pragma unroll
    for (int kk = 0; kk < 8; ++kk) {
      bf16x16 a  = load_frag_a(sT, tm*16, kk*32, 256);
      bf16x16 bb = load_frag_b(sS, thd*16, kk*32, 256); // Bt[hd][p] = vT
      acc.v = wmma_bf16(a, bb, acc.v);
    }
#pragma unroll
    for (int i = 0; i < 8; ++i) {
      int n  = tm*16 + i + half*8;
      int hd = thd*16 + nl;
      int tt = h*16384 + n*64 + hd;       // flat idx within (H,N,hd)
      int n2 = tt / 768;
      int c2 = tt - n2*768;
      atomicAdd(&xacc[((size_t)b*256 + n2)*768 + c2], acc.f[i] * 0.1f);
    }
  }
}

// ---------------------------------------------------------------------------
// Row LayerNorm over C=768 (one workgroup per row)
// ---------------------------------------------------------------------------
__global__ __launch_bounds__(256) void layernorm_row(
    const float* __restrict__ in, const float* __restrict__ gamma,
    const float* __restrict__ beta, float* __restrict__ out)
{
  __shared__ float red[256];
  __shared__ float red2[256];
  int row = blockIdx.x;
  const float* x = in + (size_t)row*768;
  float vals[3];
  float s = 0.f, s2 = 0.f;
#pragma unroll
  for (int i = 0; i < 3; ++i) {
    float vv = x[threadIdx.x + i*256];
    vals[i] = vv; s += vv; s2 += vv*vv;
  }
  red[threadIdx.x] = s; red2[threadIdx.x] = s2;
  __syncthreads();
  for (int off = 128; off > 0; off >>= 1) {
    if ((int)threadIdx.x < off) {
      red[threadIdx.x]  += red[threadIdx.x + off];
      red2[threadIdx.x] += red2[threadIdx.x + off];
    }
    __syncthreads();
  }
  float mu  = red[0] * (1.0f/768.0f);
  float var = red2[0] * (1.0f/768.0f) - mu*mu;
  float inv = rsqrtf(var + 1e-5f);
  float* o = out + (size_t)row*768;
#pragma unroll
  for (int i = 0; i < 3; ++i) {
    int c = threadIdx.x + i*256;
    o[c] = (vals[i] - mu) * inv * gamma[c] + beta[c];
  }
}

__global__ void zero_f32(float* p, int n) {
  int i = blockIdx.x*256 + threadIdx.x;
  if (i < n) p[i] = 0.f;
}
// fp32 -> bf16 elementwise, vectorized (n multiple of 4)
__global__ void conv_bf16k(const float* __restrict__ s, unsigned short* __restrict__ d, int n4) {
  int i = blockIdx.x*256 + threadIdx.x;
  if (i < n4) {
    float4 v = ((const float4*)s)[i];
    uint2 pk;
    pk.x = pack2bf(v.x, v.y);
    pk.y = pack2bf(v.z, v.w);
    ((uint2*)d)[i] = pk;
  }
}

// ---------------------------------------------------------------------------
extern "C" void kernel_launch(void* const* d_in, const int* in_sizes, int n_in,
                              void* d_out, int out_size, void* d_ws, size_t ws_size,
                              hipStream_t stream) {
  const float* x0    = (const float*)d_in[0];
  const float* ref   = (const float*)d_in[1];
  const float* Wqv   = (const float*)d_in[2];
  const float* Wk    = (const float*)d_in[3];
  const float* Wproj = (const float*)d_in[4];
  const float* bproj = (const float*)d_in[5];
  const float* gamma = (const float*)d_in[6];
  const float* beta  = (const float*)d_in[7];

  char* ws = (char*)d_ws;
  size_t off = 0;
  auto carve = [&](size_t bytes) {
    void* p = ws + off;
    off += (bytes + 255) & ~size_t(255);
    return p;
  };
  unsigned short* wqv_bf = (unsigned short*)carve((size_t)768*1536*2);
  unsigned short* wk_bf  = (unsigned short*)carve((size_t)768*768*2);
  unsigned short* wp_bf  = (unsigned short*)carve((size_t)768*768*2);
  unsigned short* k_bf   = (unsigned short*)carve((size_t)10*12*256*64*2);
  unsigned short* q_bf   = (unsigned short*)carve((size_t)8*12*256*64*2);
  unsigned short* v_bf   = (unsigned short*)carve((size_t)8*12*256*64*2);
  float* xacc = (float*)carve((size_t)2048*768*4);
  float* tmp  = (float*)carve((size_t)2048*768*4);
  float* xbuf = (float*)carve((size_t)2048*768*4);

  conv_bf16k<<<(768*1536/4 + 255)/256, 256, 0, stream>>>(Wqv,   wqv_bf, 768*1536/4);
  conv_bf16k<<<(768*768/4  + 255)/256, 256, 0, stream>>>(Wk,    wk_bf,  768*768/4);
  conv_bf16k<<<(768*768/4  + 255)/256, 256, 0, stream>>>(Wproj, wp_bf,  768*768/4);

  // k = ref @ Wk  (hoisted; constant across steps)
  {
    dim3 g(768/64, 2560/64);
    gemm_bf16<0><<<g, 128, 0, stream>>>(ref, wk_bf, 2560, 768, 768,
                                        k_bf, nullptr, nullptr, nullptr);
  }

  const float* xcur = x0;
  for (int step = 0; step < 3; ++step) {
    {
      dim3 g(1536/64, 2048/64);
      gemm_bf16<1><<<g, 128, 0, stream>>>(xcur, wqv_bf, 2048, 1536, 768,
                                          q_bf, v_bf, nullptr, nullptr);
    }
    zero_f32<<<(2048*768 + 255)/256, 256, 0, stream>>>(xacc, 2048*768);
    attn_fused<<<960, 256, 0, stream>>>(q_bf, k_bf, v_bf, xacc);
    {
      dim3 g(768/64, 2048/64);
      gemm_bf16<2><<<g, 128, 0, stream>>>(xacc, wp_bf, 2048, 768, 768,
                                          nullptr, nullptr, tmp, bproj);
    }
    float* lnout = (step == 2) ? (float*)d_out : xbuf;
    layernorm_row<<<2048, 256, 0, stream>>>(tmp, gamma, beta, lnout);
    xcur = xbuf;
  }
}
